// SpatialAttention_88356067213771
// MI455X (gfx1250) — compile-verified, matrix operand order checked by reading, not run on hardware
//
#include <hip/hip_runtime.h>

// ---------------------------------------------------------------------------
// SpatialAttention for MI455X (gfx1250, wave32, WMMA f32_16x16x32_f16 + TDM)
//
// B=64, N=512, C=64, H=8, R=17, E=21, HD=8
// Dominant math: pre_kv einsum == GEMM [8704x512]@[512x4096]  (36.5 GFLOP)
//                kv projection == GEMM [272x64]@[64x128] per (b, 16-token)
// Both run on v_wmma_f32_16x16x32_f16 with f32 accumulation.
// Stage-1 GEMM operand staging uses TENSOR_LOAD_TO_LDS (Tensor Data Mover),
// with D# LDS padding reproducing the [128][40]-half conflict-free layout.
// ---------------------------------------------------------------------------

#define BATCH 64
#define NTOK  512
#define CH    64
#define NH    8
#define RR    17
#define EE    21
#define HDIM  8
#define HR    (NH * RR)          // 136
#define MRROWS (NTOK * RR)       // 8704  rows of stage-1 GEMM
#define BCCOLS (BATCH * CH)      // 4096  cols of stage-1 GEMM
#define QSCALE 0.35355339059327373f
#define BETA   0.01f

typedef __attribute__((ext_vector_type(16))) _Float16 v16h;
typedef __attribute__((ext_vector_type(8)))  _Float16 v8h;
typedef __attribute__((ext_vector_type(4)))  _Float16 v4h;
typedef __attribute__((ext_vector_type(8)))  float    v8f;
typedef __attribute__((ext_vector_type(4)))  unsigned int u32x4;
typedef __attribute__((ext_vector_type(8)))  int          i32x8;
typedef __attribute__((ext_vector_type(4)))  int          i32x4;

union HFrag { v16h v; struct { v8h lo; v8h hi; } p; };

// Load a 16x32 f16 WMMA A/B fragment from a row-major (K-contiguous) LDS tile.
// Lane L<16 : row L,   K = [0..8) and [16..24)
// Lane L>=16: row L-16,K = [8..16) and [24..32)
__device__ __forceinline__ v16h ld_frag_lds(const _Float16* base, int lane, int stride_h) {
  const _Float16* p = base + (lane & 15) * stride_h + ((lane >> 4) << 3);
  HFrag f;
  f.p.lo = *(const v8h*)(p);
  f.p.hi = *(const v8h*)(p + 16);
  return f.v;
}

__device__ __forceinline__ v8f wmma_f16(v16h a, v16h b, v8f c) {
  return __builtin_amdgcn_wmma_f32_16x16x32_f16(
      /*neg_a=*/false, a, /*neg_b=*/false, b,
      /*c_mod=*/(short)0, c, /*reuse_a=*/false, /*reuse_b=*/false);
}

#if __has_builtin(__builtin_amdgcn_tensor_load_to_lds) && \
    __has_builtin(__builtin_amdgcn_s_wait_tensorcnt)
#define USE_TDM 1
#else
#define USE_TDM 0
#endif

#if USE_TDM
// Issue one TDM 2D tile load: global (row-major, stride_e elems of 2B) -> LDS,
// with LDS padding: every 16 DWORDs (64B tile row) insert 4 DWORDs (16B),
// i.e. LDS row pitch = 80B = 40 halves, matching the As/Bs layout.
// This toolchain's builtin is the 6-arg form:
//   (u32x4 g0, i32x8 g1, i32x4 g2, i32x4 g3, i32x8 g4, i32 cpol)
__device__ __forceinline__ void tdm_load_tile(
    unsigned lds_off, const _Float16* gptr,
    unsigned tensor_rows,      // tensor_dim1 (for OOB)
    unsigned row_len_e,        // tensor_dim0, elements
    unsigned stride_e,         // tensor_dim0_stride, elements
    unsigned tile_w_e,         // tile_dim0, elements  (32)
    unsigned tile_h)           // tile_dim1, rows      (128)
{
  unsigned long long ga = (unsigned long long)(uintptr_t)gptr;
  u32x4 g0;
  g0[0] = 1u;                                         // count=1 (valid user D#)
  g0[1] = lds_off;                                    // lds_addr (bytes)
  g0[2] = (unsigned)(ga & 0xffffffffu);               // global_addr[31:0]
  g0[3] = (unsigned)((ga >> 32) & 0x01ffffffu)        // global_addr[56:32]
        | (2u << 30);                                 // type=2 ("image")
  i32x8 g1;
  g1[0] = (int)((1u << 16)                            // data_size = 2 bytes
              | (1u << 20)                            // pad_enable
              | (3u << 22)                            // pad_interval: 16 DWORDs
              | (3u << 25));                          // pad_amount:    4 DWORDs
  g1[1] = (int)((row_len_e & 0xffffu) << 16);         // tensor_dim0[15:0]
  g1[2] = (int)(((row_len_e >> 16) & 0xffffu)         // tensor_dim0[31:16]
              | ((tensor_rows & 0xffffu) << 16));     // tensor_dim1[15:0]
  g1[3] = (int)(((tensor_rows >> 16) & 0xffffu)       // tensor_dim1[31:16]
              | ((tile_w_e & 0xffffu) << 16));        // tile_dim0
  g1[4] = (int)(tile_h & 0xffffu);                    // tile_dim1 (tile_dim2=0)
  g1[5] = (int)stride_e;                              // tensor_dim0_stride[31:0]
  g1[6] = 0;                                          // stride hi + dim1_stride lo
  g1[7] = 0;
  i32x4 z4 = {0, 0, 0, 0};
  i32x8 z8 = {0, 0, 0, 0, 0, 0, 0, 0};
  __builtin_amdgcn_tensor_load_to_lds(g0, g1, z4, z4, z8, 0);
}
#endif

// ---------------------------------------------------------------------------
// Kernel 1: per-token prologue.
//  q[b,n,:]    = (x[b,n,:] @ Wq^T) * HD^-0.5
//  bias[b,n,o] = rel_bias[h,r] + BETA * (relu(ext@W1^T+b1)@W2^T+b2)[o],
//                o = h*17+r, with mask folded in as -inf.
// ---------------------------------------------------------------------------
__global__ __launch_bounds__(64) void prep_kernel(
    const float* __restrict__ x, const float* __restrict__ ext,
    const float* __restrict__ Wq,
    const float* __restrict__ W1, const float* __restrict__ b1,
    const float* __restrict__ W2, const float* __restrict__ b2,
    const float* __restrict__ relb, const unsigned char* __restrict__ mask,
    float* __restrict__ qout, float* __restrict__ biasout) {
  __shared__ float xr[CH];
  __shared__ float er[24];
  __shared__ float hid[64];
  const long tok = blockIdx.x;          // b*512 + n
  const int  n   = (int)(tok & (NTOK - 1));
  const int  tid = threadIdx.x;

  xr[tid] = x[tok * CH + tid];
  if (tid < EE) er[tid] = ext[tok * EE + tid];
  __syncthreads();

  // q projection (scale folded in)
  float qv = 0.f;
  #pragma unroll 8
  for (int c = 0; c < CH; ++c) qv += xr[c] * Wq[tid * CH + c];
  qout[tok * CH + tid] = qv * QSCALE;

  // MLP hidden
  float hv = b1[tid];
  #pragma unroll
  for (int e = 0; e < EE; ++e) hv += er[e] * W1[tid * EE + e];
  hid[tid] = fmaxf(hv, 0.f);
  __syncthreads();

  for (int o = tid; o < HR; o += 64) {
    float s = b2[o];
    #pragma unroll 8
    for (int j = 0; j < 64; ++j) s += hid[j] * W2[o * 64 + j];
    const int h = o / RR, r = o - h * RR;
    float bias = relb[h * RR + r] + BETA * s;
    if (mask[n * RR + r]) bias = -INFINITY;
    biasout[tok * HR + o] = bias;
  }
}

// ---------------------------------------------------------------------------
// Kernel 2a/2b: pack operands f32 -> f16, K-contiguous ("NT" GEMM layout).
//  A [(m*17+r)][n] = assignment[m][n][r]
//  XT[(b*64+c)][n] = x[b][n][c]
// ---------------------------------------------------------------------------
__global__ __launch_bounds__(256) void packA_kernel(
    const float* __restrict__ assign, _Float16* __restrict__ A) {
  const long i = (long)blockIdx.x * 256 + threadIdx.x;   // over 8704*512
  if (i >= (long)MRROWS * NTOK) return;
  const long row = i >> 9;
  const int  nn  = (int)(i & (NTOK - 1));
  const long m   = row / RR;
  const int  r   = (int)(row - m * RR);
  A[i] = (_Float16)assign[(m * NTOK + nn) * RR + r];
}

__global__ __launch_bounds__(256) void packX_kernel(
    const float* __restrict__ x, _Float16* __restrict__ XT) {
  const long i = (long)blockIdx.x * 256 + threadIdx.x;   // over 4096*512
  if (i >= (long)BCCOLS * NTOK) return;
  const long row = i >> 9;
  const int  nn  = (int)(i & (NTOK - 1));
  const int  bb  = (int)(row >> 6);
  const int  c   = (int)(row & 63);
  XT[i] = (_Float16)x[((long)bb * NTOK + nn) * CH + c];
}

// ---------------------------------------------------------------------------
// Kernel 3: stage-1 GEMM  PreKV[8704][4096] = A[8704][512] * XT[4096][512]^T
// 128x128 workgroup tile, 8 waves each owning a 32x64 sub-tile (2x4 WMMA tiles).
// K staged through LDS in 32-wide slabs via the Tensor Data Mover (wave 0
// issues tensor_load_to_lds with LDS-padding descriptors; drained with
// s_wait_tensorcnt), plus global_prefetch_b8 for the next slab.
// ---------------------------------------------------------------------------
__global__ __launch_bounds__(256) void gemm1_kernel(
    const _Float16* __restrict__ A, const _Float16* __restrict__ Bt,
    _Float16* __restrict__ C) {
  __shared__ __align__(16) _Float16 As[128][40];
  __shared__ __align__(16) _Float16 Bs[128][40];

  const int tid  = threadIdx.x;
  const int lane = tid & 31;
  const int wave = tid >> 5;
  const int wm   = wave & 3;      // 4 waves along M (32 rows each)
  const int wn   = wave >> 2;     // 2 waves along N (64 cols each)
  const long arow0 = (long)blockIdx.x * 128;   // 68 tiles
  const long brow0 = (long)blockIdx.y * 128;   // 32 tiles

  v8f acc[2][4];
  v8f zero = {};
  #pragma unroll
  for (int i = 0; i < 2; ++i)
    #pragma unroll
    for (int j = 0; j < 4; ++j) acc[i][j] = zero;

#if USE_TDM
  const unsigned ldsA = (unsigned)(uintptr_t)&As[0][0];
  const unsigned ldsB = (unsigned)(uintptr_t)&Bs[0][0];
#endif

  for (int k = 0; k < NTOK; k += 32) {
#if USE_TDM
    if (wave == 0) {
      // TDM: 128-row x 32-half tiles; D# padding gives 40-half LDS pitch.
      tdm_load_tile(ldsA, &A[arow0 * NTOK + k],  MRROWS, NTOK, NTOK, 32, 128);
      tdm_load_tile(ldsB, &Bt[brow0 * NTOK + k], BCCOLS, NTOK, NTOK, 32, 128);
      __builtin_amdgcn_s_wait_tensorcnt(0);
    }
#else
    #pragma unroll
    for (int it = 0; it < 2; ++it) {
      const int s = tid + it * 256;            // 0..511
      const int row = s >> 2, seg = s & 3;
      *(uint4*)&As[row][seg * 8] =
          *(const uint4*)&A[(arow0 + row) * NTOK + k + seg * 8];
      *(uint4*)&Bs[row][seg * 8] =
          *(const uint4*)&Bt[(brow0 + row) * NTOK + k + seg * 8];
    }
#endif
    if (k + 32 < NTOK) {   // prefetch next slab (speculative, RT)
      __builtin_prefetch(&A[(arow0 + (tid >> 1)) * NTOK + k + 32], 0, 1);
      __builtin_prefetch(&Bt[(brow0 + (tid >> 1)) * NTOK + k + 32], 0, 1);
    }
    __syncthreads();

    v16h af[2], bf[4];
    #pragma unroll
    for (int i = 0; i < 2; ++i)
      af[i] = ld_frag_lds(&As[wm * 32 + i * 16][0], lane, 40);
    #pragma unroll
    for (int j = 0; j < 4; ++j)
      bf[j] = ld_frag_lds(&Bs[wn * 64 + j * 16][0], lane, 40);

    #pragma unroll
    for (int i = 0; i < 2; ++i)
      #pragma unroll
      for (int j = 0; j < 4; ++j)
        acc[i][j] = wmma_f16(af[i], bf[j], acc[i][j]);
    __syncthreads();
  }

  // C fragment: lane L, vgpr g -> (M = g + 8*(L>>4), Ncol = L&15)
  const int cn = lane & 15;
  const int mo = (lane >> 4) * 8;
  #pragma unroll
  for (int i = 0; i < 2; ++i)
    #pragma unroll
    for (int j = 0; j < 4; ++j) {
      const long r0 = arow0 + wm * 32 + i * 16 + mo;
      const long c0 = brow0 + wn * 64 + j * 16 + cn;
      #pragma unroll
      for (int g = 0; g < 8; ++g)
        C[(r0 + g) * BCCOLS + c0] = (_Float16)acc[i][j][g];
    }
}

// ---------------------------------------------------------------------------
// Kernel 4: fused tail, one workgroup per (b, 16-token tile).
//  kv[272][128] = PreKV_tile[272][64] @ Wkv^T[64][128]   (WMMA, A from global)
//  attn -> masked softmax over R=17 -> p.v -> out @ Wp^T + bp
// ---------------------------------------------------------------------------
__global__ __launch_bounds__(256) void tail_kernel(
    const _Float16* __restrict__ PreKV, const float* __restrict__ Wkv,
    const float* __restrict__ qbuf, const float* __restrict__ biasbuf,
    const float* __restrict__ Wp, const float* __restrict__ bp,
    float* __restrict__ dout) {
  __shared__ __align__(16) _Float16 kv_s[272][136];  //  73,984 B (k | v)
  __shared__ __align__(16) _Float16 wkv_s[128][72];  //  18,432 B (B operand)
  __shared__ float q_s[16][64];                      //   4,096 B
  __shared__ float bias_s[16][136];                  //   8,704 B
  __shared__ float out_s[16][64];                    //   4,096 B

  const int tid  = threadIdx.x;
  const int lane = tid & 31;
  const int wave = tid >> 5;
  const int b    = blockIdx.x >> 5;            // 0..63
  const int mt0  = (blockIdx.x & 31) * 16;     // token tile base

  // Wkv f32 -> f16 B-operand staging ([2C rows][C=K cols], K-contiguous)
  for (int s = tid; s < 2048; s += 256) {
    const int row = s >> 4, seg = s & 15;
    float4 w = *(const float4*)&Wkv[row * CH + seg * 4];
    v4h h4 = { (_Float16)w.x, (_Float16)w.y, (_Float16)w.z, (_Float16)w.w };
    *(v4h*)&wkv_s[row][seg * 4] = h4;
  }
  // q (pre-scaled) and additive bias (mask already folded as -inf)
  {
    const int m = tid >> 4, seg = tid & 15;    // 256 threads = 16x16 float4
    *(float4*)&q_s[m][seg * 4] =
        *(const float4*)&qbuf[((long)b * NTOK + mt0 + m) * CH + seg * 4];
  }
  for (int s = tid; s < 16 * HR; s += 256) {
    const int m = s / HR, o = s - m * HR;
    bias_s[m][o] = biasbuf[((long)b * NTOK + mt0 + m) * HR + o];
  }
  __syncthreads();

  // ---- WMMA: 17 M-tiles x 8 N-tiles, K = 64 (2 slabs of 32) ----
  // A fragments straight from global: rows (mt0*17 + local) x 64 halves at b*64.
  const _Float16* gA = PreKV + (long)(mt0 * RR) * BCCOLS + b * CH;
  for (int t = wave; t < 17 * 8; t += 8) {     // uniform per wave: 17 tiles each
    const int mt = t >> 3, nt = t & 7;
    const int arow = mt * 16 + (lane & 15);
    const _Float16* ap = gA + (long)arow * BCCOLS + ((lane >> 4) << 3);
    v8f acc = {};
    #pragma unroll
    for (int kc = 0; kc < 2; ++kc) {
      HFrag af;
      af.p.lo = *(const v8h*)(ap + kc * 32);
      af.p.hi = *(const v8h*)(ap + kc * 32 + 16);
      v16h bf = ld_frag_lds(&wkv_s[nt * 16][kc * 32], lane, 72);
      acc = wmma_f16(af.v, bf, acc);
    }
    const int cn = lane & 15;
    const int mo = (lane >> 4) * 8;
    #pragma unroll
    for (int g = 0; g < 8; ++g)
      kv_s[mt * 16 + mo + g][nt * 16 + cn] = (_Float16)acc[g];
  }
  __syncthreads();

  // ---- attention: 1xR per (m,h), masked softmax, p.v ----
  if (tid < 128) {
    const int m = tid >> 3, h = tid & 7;
    float p[RR];
    float mx = -INFINITY;
    #pragma unroll
    for (int r = 0; r < RR; ++r) {
      float s = bias_s[m][h * RR + r];
      float dot = 0.f;
      #pragma unroll
      for (int d = 0; d < HDIM; ++d)
        dot += q_s[m][h * HDIM + d] * (float)kv_s[m * RR + r][h * HDIM + d];
      s += dot;                                // -inf stays -inf
      p[r] = s;
      mx = fmaxf(mx, s);
    }
    float sum = 0.f;
    #pragma unroll
    for (int r = 0; r < RR; ++r) { float e = __expf(p[r] - mx); p[r] = e; sum += e; }
    const float inv = 1.f / sum;
    #pragma unroll
    for (int d = 0; d < HDIM; ++d) {
      float o = 0.f;
      #pragma unroll
      for (int r = 0; r < RR; ++r)
        o += p[r] * (float)kv_s[m * RR + r][CH + h * HDIM + d];
      out_s[m][h * HDIM + d] = o * inv;
    }
  }
  __syncthreads();

  // ---- final projection: y = out @ Wp^T + bp ----
  #pragma unroll
  for (int it = 0; it < 4; ++it) {
    const int s = tid + it * 256;              // 16*64 = 1024 outputs
    const int m = s >> 6, co = s & 63;
    float y = bp[co];
    #pragma unroll 8
    for (int c = 0; c < CH; ++c) y += out_s[m][c] * Wp[co * CH + c];
    dout[((long)b * NTOK + mt0 + m) * CH + co] = y;
  }
}

// ---------------------------------------------------------------------------
// Host launcher. Workspace layout (bytes, ~110.6 MB total):
//   q      f32 [B,N,64]        @ 0           (8,388,608)
//   bias   f32 [B,N,136]       @ 8,388,608   (17,825,792)
//   A      f16 [8704,512]      @ 26,214,400  (8,912,896)
//   XT     f16 [4096,512]      @ 35,127,296  (4,194,304)
//   PreKV  f16 [8704,4096]     @ 39,321,600  (71,303,168)
// ---------------------------------------------------------------------------
extern "C" void kernel_launch(void* const* d_in, const int* in_sizes, int n_in,
                              void* d_out, int out_size, void* d_ws, size_t ws_size,
                              hipStream_t stream) {
  const float* x      = (const float*)d_in[0];
  const float* ext    = (const float*)d_in[1];
  const float* assign = (const float*)d_in[2];
  const float* Wq     = (const float*)d_in[3];
  const float* Wkv    = (const float*)d_in[4];
  const float* relb   = (const float*)d_in[5];
  const float* W1     = (const float*)d_in[6];
  const float* b1     = (const float*)d_in[7];
  const float* W2     = (const float*)d_in[8];
  const float* b2     = (const float*)d_in[9];
  const float* Wp     = (const float*)d_in[10];
  const float* bp     = (const float*)d_in[11];
  const unsigned char* mask = (const unsigned char*)d_in[12];

  char* ws = (char*)d_ws;
  float*    qbuf    = (float*)(ws);
  float*    biasbuf = (float*)(ws + 8388608);
  _Float16* Af      = (_Float16*)(ws + 26214400);
  _Float16* XT      = (_Float16*)(ws + 35127296);
  _Float16* PreKV   = (_Float16*)(ws + 39321600);

  prep_kernel<<<BATCH * NTOK, 64, 0, stream>>>(x, ext, Wq, W1, b1, W2, b2,
                                               relb, mask, qbuf, biasbuf);
  packA_kernel<<<(MRROWS * NTOK) / 256, 256, 0, stream>>>(assign, Af);
  packX_kernel<<<(BCCOLS * NTOK) / 256, 256, 0, stream>>>(x, XT);
  gemm1_kernel<<<dim3(MRROWS / 128, BCCOLS / 128), 256, 0, stream>>>(Af, XT, PreKV);
  tail_kernel<<<BATCH * (NTOK / 16), 256, 0, stream>>>(PreKV, Wkv, qbuf, biasbuf,
                                                       Wp, bp, (float*)d_out);
}
